// MPNN_LSTM_80719615361183
// MI455X (gfx1250) — matrix operand order repaired and checked by reading, not compiled, not run on hardware
//
#include <hip/hip_runtime.h>
#include <hip/hip_bf16.h>

// ---------------------------------------------------------------------------
// MPNN-LSTM forward on gfx1250 (CDNA5). Dense GEMMs via v_wmma_f32_16x16x32_bf16
// (f32 accumulate) with fragment-native packed B, graph scatter via f32 atomics,
// BN/LSTM pointwise.
// ---------------------------------------------------------------------------

#define NN  50000
#define EE  800000
#define IND 256
#define HD  256

typedef __attribute__((ext_vector_type(16))) __bf16 v16bf;
typedef __attribute__((ext_vector_type(8)))  float  v8f;

static __device__ __forceinline__ unsigned short f2bf(float f) {
    unsigned int u = __float_as_uint(f);
    // round-to-nearest-even fp32 -> bf16
    u = u + 0x7FFFu + ((u >> 16) & 1u);
    return (unsigned short)(u >> 16);
}
static __device__ __forceinline__ float sigmoidf(float x) {
    return 1.0f / (1.0f + __expf(-x));
}

// ------------------------------- tiny utility kernels ----------------------
__global__ void k_fill(float* p, float v, int n) {
    int i = blockIdx.x * blockDim.x + threadIdx.x;
    if (i < n) p[i] = v;
}
__global__ void k_fill_bias(float* __restrict__ agg, const float* __restrict__ b, int n) {
    int i = blockIdx.x * blockDim.x + threadIdx.x;   // over n*HD
    if (i < n * HD) agg[i] = b[i & (HD - 1)];
}
__global__ void k_deg_scatter(const int* __restrict__ dst, const float* __restrict__ ew,
                              float* __restrict__ deg, int e) {
    int i = blockIdx.x * blockDim.x + threadIdx.x;
    if (i < e) atomicAdd(&deg[dst[i]], ew[i]);
}
__global__ void k_rsqrt(float* __restrict__ d, int n) {
    int i = blockIdx.x * blockDim.x + threadIdx.x;
    if (i < n) { float v = d[i]; d[i] = (v > 0.0f) ? rsqrtf(v) : 0.0f; }
}
__global__ void k_cvt_bf16(const float* __restrict__ s, unsigned short* __restrict__ d, int n) {
    int i = blockIdx.x * blockDim.x + threadIdx.x;
    if (i < n) d[i] = f2bf(s[i]);
}

// Pack weight matrix into WMMA-fragment-native bf16 layout.
// Logical B is [K,M]; source is src[k*M+m] or (transpose) src[m*K+k].
// Packed: for K-tile kt (32 rows) and N-tile nt (16 cols):
//   lane l = (half=k16)*16 + n, element i = k&15 :
//   Bp[ ((kt*(M/16)+nt)*32 + l)*16 + i ] = B[kt*32 + half*16 + i][nt*16 + n]
// -> a wave's B fragment = 32 lanes x 16 contiguous bf16 (32B/lane, coalesced).
__global__ void k_pack_B(const float* __restrict__ src, unsigned short* __restrict__ dstp,
                         int K, int M, int transpose) {
    int o = blockIdx.x * blockDim.x + threadIdx.x;
    if (o >= K * M) return;
    int i    = o & 15;
    int l    = (o >> 4) & 31;
    int tile = o >> 9;
    int ntiles = M >> 4;
    int nt = tile % ntiles;
    int kt = tile / ntiles;
    int half = l >> 4, n = l & 15;
    int k = kt * 32 + half * 16 + i;
    int m = nt * 16 + n;
    float v = transpose ? src[(size_t)m * K + k] : src[(size_t)k * M + m];
    dstp[o] = f2bf(v);
}

// ------------------------------- WMMA bf16 GEMM -----------------------------
// C[nrows,M] = A[nrows,K](bf16,lda) @ B[K,M](packed bf16) (+bias0+bias1), C fp32.
// Block: 256 thr = 8 waves -> 128x128 tile; wave = 16x128 (8 wmma accumulators).
// Scheduling groups pin each K-step to: 2 DS reads (A frag) -> 16 VMEM reads
// (all B frags) -> 8 back-to-back WMMAs (independent accumulators, one wait).
#define TM 128
#define TN 128
#define TK 32
#define APAD 40   // As row stride (bf16): 80 bytes, 16B-aligned, conflict-free

__global__ __launch_bounds__(256)
void k_gemm_bf16(const unsigned short* __restrict__ A, int lda, int nrows, int K,
                 const unsigned short* __restrict__ Bp, int M,
                 float* __restrict__ C, int ldc,
                 const float* __restrict__ bias0, const float* __restrict__ bias1)
{
    __shared__ unsigned short As[TM][APAD];     // 128 x 40 bf16 = 10 KB

    const int tid  = threadIdx.x;
    const int wid  = tid >> 5;                  // 0..7 -> 16-row slice
    const int lane = tid & 31;
    const int half = lane >> 4;
    const int l16  = lane & 15;

    const int row0 = blockIdx.y * TM;
    const int col0 = blockIdx.x * TN;
    const int ntiles = M >> 4;
    const int nt0 = col0 >> 4;
    const bool fullTile = (row0 + TM) <= nrows;   // block-uniform

    v8f zero = {};
    v8f acc[8];
    #pragma unroll
    for (int s = 0; s < 8; ++s) acc[s] = zero;

    for (int k0 = 0; k0 < K; k0 += TK) {
        // --- stage A tile (128x32 bf16): 2 x b128 per thread, b128 LDS stores
        if (fullTile) {
            #pragma unroll
            for (int j = 0; j < 2; ++j) {
                int c   = tid + j * 256;        // 0..511
                int r   = c >> 2;               // 0..127
                int seg = (c & 3) * 8;          // 0,8,16,24
                uint4 val = *(const uint4*)(A + (size_t)(row0 + r) * lda + (k0 + seg));
                *(uint4*)&As[r][seg] = val;
            }
        } else {
            #pragma unroll
            for (int j = 0; j < 2; ++j) {
                int c   = tid + j * 256;
                int r   = c >> 2;
                int seg = (c & 3) * 8;
                int grow = row0 + r;
                uint4 val;
                if (grow < nrows) {
                    val = *(const uint4*)(A + (size_t)grow * lda + (k0 + seg));
                } else {
                    val.x = val.y = val.z = val.w = 0u;
                }
                *(uint4*)&As[r][seg] = val;
            }
        }
        __syncthreads();

        // --- A fragment (16x32 bf16, k = half*16 + i, matches packed B): two ds_load_b128
        union { v16bf v; uint4 q[2]; } af;
        {
            int r = wid * 16 + l16;
            af.q[0] = *(const uint4*)&As[r][half * 16];
            af.q[1] = *(const uint4*)&As[r][half * 16 + 8];
        }

        const int kt = k0 >> 5;
        // prefetch next K-step's packed-B slice
        if (k0 + TK < K) {
            const unsigned short* nb =
                Bp + ((((size_t)(kt + 1) * ntiles + nt0) * 32) << 4) + (tid << 3);
            __builtin_prefetch(nb, 0, 0);
        }

        // --- preload all 8 B fragments (16 x global_load_b128, consecutive),
        //     then 8 back-to-back WMMAs on independent accumulators
        union { v16bf v; uint4 q[2]; } bfm[8];
        {
            const uint4* bp =
                (const uint4*)(Bp + ((((size_t)kt * ntiles + nt0) * 32 + lane) << 4));
            #pragma unroll
            for (int s = 0; s < 8; ++s) {
                bfm[s].q[0] = bp[s * 64];
                bfm[s].q[1] = bp[s * 64 + 1];
            }
        }
        #pragma unroll
        for (int s = 0; s < 8; ++s) {
            acc[s] = __builtin_amdgcn_wmma_f32_16x16x32_bf16(
                         false, af.v, false, bfm[s].v, (short)0, acc[s], false, false);
        }

#if defined(__has_builtin)
#if __has_builtin(__builtin_amdgcn_sched_group_barrier)
        // Pin per-iteration schedule: DS reads (A frag), then all VMEM reads
        // (B frags), then the 8-WMMA burst. Masks: 0x100=DS read, 0x020=VMEM
        // read, 0x008=MFMA/WMMA.
        __builtin_amdgcn_sched_group_barrier(0x100, 2, 0);
        __builtin_amdgcn_sched_group_barrier(0x020, 16, 0);
        __builtin_amdgcn_sched_group_barrier(0x008, 8, 0);
#endif
#endif
        __syncthreads();
    }

    // --- store C (VGPR p: lanes0-15 -> M=p, lanes16-31 -> M=p+8)
    #pragma unroll
    for (int s = 0; s < 8; ++s) {
        int col = col0 + s * 16 + l16;
        float badd = 0.0f;
        if (bias0) badd += bias0[col];
        if (bias1) badd += bias1[col];
        #pragma unroll
        for (int p = 0; p < 8; ++p) {
            int row = row0 + wid * 16 + p + half * 8;
            if (row < nrows) C[(size_t)row * ldc + col] = acc[s][p] + badd;
        }
    }
}

// ------------------------------- message scatter ----------------------------
// edges [0,E): (src,dst,ew); edges [E,E+N): self loops weight 1. 64 groups of
// 4 channels per edge; f32 atomics into agg.
__global__ void k_scatter(const float* __restrict__ xw, float* __restrict__ agg,
                          const int* __restrict__ src, const int* __restrict__ dst,
                          const float* __restrict__ ew, const float* __restrict__ dinv,
                          int E, int n)
{
    int idx = blockIdx.x * blockDim.x + threadIdx.x;     // (E+n)*64 total
    int total = (E + n) * 64;
    if (idx >= total) return;
    int e = idx >> 6;
    int g = idx & 63;
    int s, d; float w;
    if (e < E) { s = src[e]; d = dst[e]; w = ew[e]; }
    else       { s = e - E; d = s;       w = 1.0f;  }
    float nrm = dinv[s] * w * dinv[d];
    const float4 m = *(const float4*)(xw + (size_t)s * HD + g * 4);
    float* out = agg + (size_t)d * HD + g * 4;
    atomicAdd(out + 0, m.x * nrm);
    atomicAdd(out + 1, m.y * nrm);
    atomicAdd(out + 2, m.z * nrm);
    atomicAdd(out + 3, m.w * nrm);
}

// ------------------------------- BatchNorm (relu fused) ---------------------
#define BN_ROWS 128
__global__ __launch_bounds__(HD)
void k_bn_stats(const float* __restrict__ agg, float* __restrict__ sum,
                float* __restrict__ sumsq, int n)
{
    int c = threadIdx.x;                   // channel
    int r0 = blockIdx.x * BN_ROWS;
    float s = 0.0f, s2 = 0.0f;
    for (int r = 0; r < BN_ROWS; ++r) {
        int row = r0 + r;
        if (row < n) {
            float v = fmaxf(agg[(size_t)row * HD + c], 0.0f);
            s += v; s2 += v * v;
        }
    }
    atomicAdd(&sum[c], s);
    atomicAdd(&sumsq[c], s2);
}

__global__ void k_bn_apply(const float* __restrict__ agg, const float* __restrict__ sum,
                           const float* __restrict__ sumsq,
                           const float* __restrict__ gamma, const float* __restrict__ beta,
                           unsigned short* __restrict__ xc, int colOff, int n)
{
    int i = blockIdx.x * blockDim.x + threadIdx.x;       // n*HD
    if (i >= n * HD) return;
    int row = i >> 8, c = i & 255;
    float inv_n = 1.0f / (float)n;
    float mu  = sum[c] * inv_n;
    float var = sumsq[c] * inv_n - mu * mu;
    float v = fmaxf(agg[i], 0.0f);
    float y = (v - mu) * rsqrtf(var + 1e-5f) * gamma[c] + beta[c];
    xc[(size_t)row * (2 * HD) + colOff + c] = f2bf(y);
}

// ------------------------------- LSTM pointwise -----------------------------
// gates [n,4H] = (i,f,g,o); h0=c0=0 -> c = sig(i)*tanh(g); h = sig(o)*tanh(c)
__global__ void k_lstm(const float* __restrict__ gates, float* __restrict__ out,
                       int outOff, unsigned short* __restrict__ hb, int n)
{
    int i = blockIdx.x * blockDim.x + threadIdx.x;       // n*HD
    if (i >= n * HD) return;
    int row = i >> 8, u = i & 255;
    const float* g = gates + (size_t)row * (4 * HD);
    float gi = g[u], gg = g[2 * HD + u], go = g[3 * HD + u];
    float c = sigmoidf(gi) * tanhf(gg);
    float h = sigmoidf(go) * tanhf(c);
    out[(size_t)row * (2 * HD + IND) + outOff + u] = h;
    if (hb) hb[(size_t)row * HD + u] = f2bf(h);
}

__global__ void k_copy_x(const float* __restrict__ x, float* __restrict__ out, int n) {
    int i = blockIdx.x * blockDim.x + threadIdx.x;       // n*IND
    if (i >= n * IND) return;
    int row = i >> 8, c = i & 255;
    out[(size_t)row * (2 * HD + IND) + 2 * HD + c] = x[i];
}

// ------------------------------- host orchestration -------------------------
extern "C" void kernel_launch(void* const* d_in, const int* in_sizes, int n_in,
                              void* d_out, int out_size, void* d_ws, size_t ws_size,
                              hipStream_t stream) {
    (void)in_sizes; (void)n_in; (void)out_size; (void)ws_size;
    const float* x     = (const float*)d_in[0];
    const int*   ei    = (const int*)  d_in[1];
    const float* ew    = (const float*)d_in[2];
    const float* W1    = (const float*)d_in[3];
    const float* b1    = (const float*)d_in[4];
    const float* W2    = (const float*)d_in[5];
    const float* b2    = (const float*)d_in[6];
    const float* gm1   = (const float*)d_in[7];
    const float* bt1   = (const float*)d_in[8];
    const float* gm2   = (const float*)d_in[9];
    const float* bt2   = (const float*)d_in[10];
    const float* Wih1  = (const float*)d_in[11];
    const float* bih1  = (const float*)d_in[13];
    const float* bhh1  = (const float*)d_in[14];
    const float* Wih2  = (const float*)d_in[15];
    const float* bih2  = (const float*)d_in[17];
    const float* bhh2  = (const float*)d_in[18];
    const int* src = ei;
    const int* dst = ei + EE;
    float* out = (float*)d_out;

    char* ws = (char*)d_ws;
    // overlaid workspace layout (bytes)
    size_t offGates = 0;                                   // [N,1024] f32 (also x_bf16 early)
    size_t offXw    = offGates + (size_t)NN * 1024 * 4;    // [N,256]  f32 (also hL1 bf16 late)
    size_t offAgg   = offXw    + (size_t)NN * HD * 4;      // [N,256]  f32
    size_t offXc    = offAgg   + (size_t)NN * HD * 4;      // [N,512]  bf16
    size_t offW1b   = offXc    + (size_t)NN * 2 * HD * 2;
    size_t offW2b   = offW1b   + (size_t)IND * HD * 2;
    size_t offWih1t = offW2b   + (size_t)HD * HD * 2;
    size_t offWih2t = offWih1t + (size_t)(2 * HD) * (4 * HD) * 2;
    size_t offDeg   = offWih2t + (size_t)HD * (4 * HD) * 2;
    size_t offStats = offDeg   + ((size_t)NN * 4 + 255) / 256 * 256;

    float*          gates = (float*)(ws + offGates);
    unsigned short* xb    = (unsigned short*)(ws + offGates);   // overlay
    float*          xw    = (float*)(ws + offXw);
    unsigned short* hL1b  = (unsigned short*)(ws + offXw);      // overlay
    float*          agg   = (float*)(ws + offAgg);
    unsigned short* Xc    = (unsigned short*)(ws + offXc);
    unsigned short* W1b   = (unsigned short*)(ws + offW1b);
    unsigned short* W2b   = (unsigned short*)(ws + offW2b);
    unsigned short* Wih1t = (unsigned short*)(ws + offWih1t);
    unsigned short* Wih2t = (unsigned short*)(ws + offWih2t);
    float*          deg   = (float*)(ws + offDeg);
    float*          bsum  = (float*)(ws + offStats);
    float*          bsq   = bsum + HD;

    const int B = 256;
    const int nh   = NN * HD;                 // 12.8M
    const int gNH  = (nh + B - 1) / B;
    const int gN   = (NN + B - 1) / B;
    const int gE   = (EE + B - 1) / B;
    const int gSct = ((EE + NN) * 64 + B - 1) / B;
    const int gBN  = (NN + BN_ROWS - 1) / BN_ROWS;
    dim3 gemm256((HD) / TN, (NN + TM - 1) / TM);       // 2 x 391
    dim3 gemm1024((4 * HD) / TN, (NN + TM - 1) / TM);  // 8 x 391

    // ---- GCN normalization: deg = 1 (self loop) + sum_e w[dst]; dinv = rsqrt
    k_fill<<<gN, B, 0, stream>>>(deg, 1.0f, NN);
    k_deg_scatter<<<gE, B, 0, stream>>>(dst, ew, deg, EE);
    k_rsqrt<<<gN, B, 0, stream>>>(deg, NN);

    // ---- weight packing (WMMA-native bf16) / input conversion
    k_pack_B<<<(IND * HD + B - 1) / B, B, 0, stream>>>(W1, W1b, IND, HD, 0);
    k_pack_B<<<(HD * HD + B - 1) / B, B, 0, stream>>>(W2, W2b, HD, HD, 0);
    k_pack_B<<<(2 * HD * 4 * HD + B - 1) / B, B, 0, stream>>>(Wih1, Wih1t, 2 * HD, 4 * HD, 1);
    k_pack_B<<<(HD * 4 * HD + B - 1) / B, B, 0, stream>>>(Wih2, Wih2t, HD, 4 * HD, 1);
    k_cvt_bf16<<<(NN * IND + B - 1) / B, B, 0, stream>>>(x, xb, NN * IND);

    // ---- GCN layer 1: xw = x @ W1 ; agg = b1 + scatter ; BN -> Xc[:, :256]
    k_gemm_bf16<<<gemm256, B, 0, stream>>>(xb, IND, NN, IND, W1b, HD, xw, HD, nullptr, nullptr);
    k_fill_bias<<<gNH, B, 0, stream>>>(agg, b1, NN);
    k_scatter<<<gSct, B, 0, stream>>>(xw, agg, src, dst, ew, deg, EE, NN);
    k_fill<<<(2 * HD + B - 1) / B, B, 0, stream>>>(bsum, 0.0f, 2 * HD);
    k_bn_stats<<<gBN, HD, 0, stream>>>(agg, bsum, bsq, NN);
    k_bn_apply<<<gNH, B, 0, stream>>>(agg, bsum, bsq, gm1, bt1, Xc, 0, NN);

    // ---- GCN layer 2: h1(bf16, Xc cols 0:256) @ W2 ; scatter ; BN -> Xc[:,256:]
    k_gemm_bf16<<<gemm256, B, 0, stream>>>(Xc, 2 * HD, NN, HD, W2b, HD, xw, HD, nullptr, nullptr);
    k_fill_bias<<<gNH, B, 0, stream>>>(agg, b2, NN);
    k_scatter<<<gSct, B, 0, stream>>>(xw, agg, src, dst, ew, deg, EE, NN);
    k_fill<<<(2 * HD + B - 1) / B, B, 0, stream>>>(bsum, 0.0f, 2 * HD);
    k_bn_stats<<<gBN, HD, 0, stream>>>(agg, bsum, bsq, NN);
    k_bn_apply<<<gNH, B, 0, stream>>>(agg, bsum, bsq, gm2, bt2, Xc, HD, NN);

    // ---- LSTM 1: gates = Xc @ Wih1^T + bih1 + bhh1 ; pointwise -> out[:, :256]
    k_gemm_bf16<<<gemm1024, B, 0, stream>>>(Xc, 2 * HD, NN, 2 * HD, Wih1t, 4 * HD,
                                            gates, 4 * HD, bih1, bhh1);
    k_lstm<<<gNH, B, 0, stream>>>(gates, out, 0, hL1b, NN);

    // ---- LSTM 2: gates = hL1 @ Wih2^T + bih2 + bhh2 ; pointwise -> out[:,256:512]
    k_gemm_bf16<<<gemm1024, B, 0, stream>>>(hL1b, HD, NN, HD, Wih2t, 4 * HD,
                                            gates, 4 * HD, bih2, bhh2);
    k_lstm<<<gNH, B, 0, stream>>>(gates, out, HD, nullptr, NN);

    // ---- output tail: out[:, 512:768] = x
    k_copy_x<<<(NN * IND + B - 1) / B, B, 0, stream>>>(x, out, NN);
}
